// LSSVolume_85968065397287
// MI455X (gfx1250) — compile-verified, matrix operand order checked by reading, not run on hardware
//
#include <hip/hip_runtime.h>

typedef __attribute__((ext_vector_type(16))) _Float16 v16h;
typedef __attribute__((ext_vector_type(8)))  float    v8f;
typedef __attribute__((ext_vector_type(2)))  float    v2f;

#ifndef __has_builtin
#define __has_builtin(x) 0
#endif
#if __has_builtin(__builtin_amdgcn_wmma_f32_16x16x4_f32)
#define USE_F32_WMMA 1
#else
#define USE_F32_WMMA 0
#endif

#define BN_   2
#define CCH   64
#define DD    40
#define HH    64
#define WW    128
#define NXX   128
#define NXY   128
#define NXZ   16
#define HW    (HH*WW)          /* 8192  */
#define CH_STRIDE (NXX*NXY)    /* 16384 */
#define PIXELS (BN_*HH*WW)     /* 16384 */
#define WAVES_PER_BLOCK 4

// ---------------------------------------------------------------------------
// Kernel 1: zero the output grid (134 MB) with b128 stores.
// ---------------------------------------------------------------------------
__global__ __launch_bounds__(256) void lss_zero_kernel(float4* __restrict__ out4,
                                                       long n4) {
    long i = (long)blockIdx.x * blockDim.x + threadIdx.x;
    if (i < n4) {
        out4[i] = make_float4(0.f, 0.f, 0.f, 0.f);
    }
}

// ---------------------------------------------------------------------------
// Kernel 2: per-pixel softmax + WMMA rank-1 lift + atomic scatter (splat).
// One wave32 per pixel; 4 waves per block.
// ---------------------------------------------------------------------------
__global__ __launch_bounds__(32*WAVES_PER_BLOCK) void lss_splat_kernel(
    const float* __restrict__ img_feats,     // [BN, C=64, H, W]
    const float* __restrict__ depth_logits,  // [BN, D=40, H, W]
    const float* __restrict__ geom,          // [B, 1, D, H, W, 3]
    float* __restrict__ out)                 // [B, Z*C, X, Y] = [B,Z,C,X,Y]
{
    __shared__ float s_depth[WAVES_PER_BLOCK][48]; // softmax weights (48: pad M-tile 2)
    __shared__ int   s_base [WAVES_PER_BLOCK][48]; // per-d output base idx, -1 = culled

    const int tid   = threadIdx.x;
    const int wv    = tid >> 5;        // wave in block
    const int lane  = tid & 31;
    const int pixel = blockIdx.x * WAVES_PER_BLOCK + wv;

    const int bn  = pixel >> 13;       // / (H*W)
    const int rem = pixel & (HW - 1);
    const int h   = rem >> 7;          // / W
    const int px  = rem & (WW - 1);

    // ---- softmax over D=40 depth bins (wave32 reduction) ----
    const int dlbase = bn * (DD * HW) + h * WW + px;
    float x1 = depth_logits[dlbase + lane * HW];                            // d = lane
    float x2 = (lane < 8) ? depth_logits[dlbase + (lane + 32) * HW] : -3.4e38f; // d = lane+32
    float m = fmaxf(x1, x2);
    #pragma unroll
    for (int off = 16; off > 0; off >>= 1) m = fmaxf(m, __shfl_xor(m, off, 32));
    float e1 = __expf(x1 - m);
    float e2 = (lane < 8) ? __expf(x2 - m) : 0.f;
    float s = e1 + e2;
    #pragma unroll
    for (int off = 16; off > 0; off >>= 1) s += __shfl_xor(s, off, 32);
    const float inv = 1.f / s;

    s_depth[wv][lane] = e1 * inv;
    if (lane < 8)                s_depth[wv][lane + 32] = e2 * inv;
    if (lane >= 8 && lane < 16)  s_depth[wv][lane + 32] = 0.f;   // pad rows 40..47

    // ---- per-depth voxel coordinate -> linear output base (or -1) ----
    {
        const int d  = lane;
        const int gi = (((bn * DD + d) * HH + h) * WW + px) * 3;
        const float gx = geom[gi + 0], gy = geom[gi + 1], gz = geom[gi + 2];
        const int cx = (int)floorf((gx -   0.0f) / 0.4f);
        const int cy = (int)floorf((gy +  25.6f) / 0.4f);
        const int cz = (int)floorf((gz +   2.0f) / 0.4f);
        const bool kept = (cx >= 0) & (cx < NXX) & (cy >= 0) & (cy < NXY) &
                          (cz >= 0) & (cz < NXZ);
        s_base[wv][d] = kept ? ((bn * NXZ + cz) * (CCH * CH_STRIDE) + cx * NXY + cy) : -1;
    }
    if (lane < 8) {
        const int d  = lane + 32;
        const int gi = (((bn * DD + d) * HH + h) * WW + px) * 3;
        const float gx = geom[gi + 0], gy = geom[gi + 1], gz = geom[gi + 2];
        const int cx = (int)floorf((gx -   0.0f) / 0.4f);
        const int cy = (int)floorf((gy +  25.6f) / 0.4f);
        const int cz = (int)floorf((gz +   2.0f) / 0.4f);
        const bool kept = (cx >= 0) & (cx < NXX) & (cy >= 0) & (cy < NXY) &
                          (cz >= 0) & (cz < NXZ);
        s_base[wv][d] = kept ? ((bn * NXZ + cz) * (CCH * CH_STRIDE) + cx * NXY + cy) : -1;
    }

    __syncthreads();

    const int fbase = bn * (CCH * HW) + h * WW + px;
    const int mhi   = (lane >> 4) << 3;   // C/D rows: M=r (lanes 0-15) or M=r+8 (16-31)
    const int nlo   = lane & 15;          // C/D col:  N = lane & 15

#if USE_F32_WMMA
    // ======== exact fp32 path: V_WMMA_F32_16X16X4_F32, K=0 row only ========
    // A (16x4 f32, 2 VGPRs): lanes 0-15 hold K=0 (elem 0) / K=1 (elem 1);
    // lanes 16-31 hold K=2/K=3.  Only K=0 nonzero: elem 0, lanes 0-15, M=lane.
    // B (4x16 f32, 2 VGPRs): elem 0 = row K=0 (lanes 0-15, N=lane) / K=2 (16-31).
    v2f bfr[4];
    #pragma unroll
    for (int u = 0; u < 4; ++u) {
        v2f b; b[0] = 0.f; b[1] = 0.f;
        if (lane < 16) b[0] = img_feats[fbase + (u * 16 + lane) * HW];
        bfr[u] = b;
    }

    #pragma unroll
    for (int t = 0; t < 3; ++t) {
        v2f a; a[0] = 0.f; a[1] = 0.f;
        if (lane < 16) a[0] = s_depth[wv][t * 16 + lane];

        // preload this tile's 8 output bases (fold the d<40 pad guard in)
        int bofs[8];
        #pragma unroll
        for (int r = 0; r < 8; ++r) {
            const int drow = t * 16 + mhi + r;
            bofs[r] = (drow < DD) ? s_base[wv][drow] : -1;
        }

        #pragma unroll
        for (int u = 0; u < 4; ++u) {
            v8f acc;
            #pragma unroll
            for (int i = 0; i < 8; ++i) acc[i] = 0.f;
            acc = __builtin_amdgcn_wmma_f32_16x16x4_f32(
                /*neg_a=*/false, a, /*neg_b=*/false, bfr[u],
                /*c_mod=*/(short)0, acc, /*reuse_a=*/false, /*reuse_b=*/false);

            const unsigned chofs = (unsigned)(u * 16 + nlo) * CH_STRIDE;
            #pragma unroll
            for (int r = 0; r < 8; ++r) {
                if (bofs[r] >= 0)
                    atomicAdd(out + (chofs + (unsigned)bofs[r]), acc[r]);
            }
        }
    }
#else
    // ======== fallback: confirmed V_WMMA_F32_16X16X32_F16, K=0 row only ====
    v16h bfr[4];
    #pragma unroll
    for (int u = 0; u < 4; ++u) {
        v16h b;
        #pragma unroll
        for (int i = 0; i < 16; ++i) b[i] = (_Float16)0.f;
        if (lane < 16) b[0] = (_Float16)img_feats[fbase + (u * 16 + lane) * HW];
        bfr[u] = b;
    }

    #pragma unroll
    for (int t = 0; t < 3; ++t) {
        v16h a;
        #pragma unroll
        for (int i = 0; i < 16; ++i) a[i] = (_Float16)0.f;
        if (lane < 16) a[0] = (_Float16)s_depth[wv][t * 16 + lane];

        int bofs[8];
        #pragma unroll
        for (int r = 0; r < 8; ++r) {
            const int drow = t * 16 + mhi + r;
            bofs[r] = (drow < DD) ? s_base[wv][drow] : -1;
        }

        #pragma unroll
        for (int u = 0; u < 4; ++u) {
            v8f acc;
            #pragma unroll
            for (int i = 0; i < 8; ++i) acc[i] = 0.f;
            acc = __builtin_amdgcn_wmma_f32_16x16x32_f16(
                /*neg_a=*/false, a, /*neg_b=*/false, bfr[u],
                /*c_mod=*/(short)0, acc, /*reuse_a=*/false, /*reuse_b=*/false);

            const unsigned chofs = (unsigned)(u * 16 + nlo) * CH_STRIDE;
            #pragma unroll
            for (int r = 0; r < 8; ++r) {
                if (bofs[r] >= 0)
                    atomicAdd(out + (chofs + (unsigned)bofs[r]), acc[r]);
            }
        }
    }
#endif
}

// ---------------------------------------------------------------------------
extern "C" void kernel_launch(void* const* d_in, const int* in_sizes, int n_in,
                              void* d_out, int out_size, void* d_ws, size_t ws_size,
                              hipStream_t stream) {
    const float* img_feats    = (const float*)d_in[0];
    const float* depth_logits = (const float*)d_in[1];
    const float* geom         = (const float*)d_in[2];
    float* out = (float*)d_out;

    // 1) zero the voxel grid (out_size = 2*1024*128*128, divisible by 4)
    const long n4 = (long)out_size / 4;
    const int zb = (int)((n4 + 255) / 256);
    lss_zero_kernel<<<zb, 256, 0, stream>>>((float4*)out, n4);

    // 2) lift + splat: one wave32 per pixel
    lss_splat_kernel<<<PIXELS / WAVES_PER_BLOCK, 32 * WAVES_PER_BLOCK, 0, stream>>>(
        img_feats, depth_logits, geom, out);
}
